// SingleGateHybridUnit_81217831567454
// MI455X (gfx1250) — matrix operand order repaired and checked by reading, not compiled, not run on hardware
//
#include <hip/hip_runtime.h>
#include <hip/hip_bf16.h>
#include <stdint.h>

// ---------------------------------------------------------------------------
// SingleGateHybridUnit on MI455X (gfx1250, wave32, WMMA + TDM)
//
// Phase 1 (parallel, tiny): H0 = ss(Xv@W0 + b0), pd = Xv@Pd + bd,
//                           pz = Xv@Pz + bz            (fp32 VALU kernel)
// Phase 2 (persistent):     1024-step recurrence. 32 persistent WGs, each
//   owns 16 columns of U. Weights cached in LDS as bf16 (N-major); state H
//   ping-pongs through global as bf16 (fp32 master copy lives in registers
//   of the owning wave). v_wmma_f32_16x16x32_bf16 does all matmuls.
//   H global->LDS staging uses the Tensor Data Mover (tensor_load_to_lds,
//   TENSORcnt), issued by wave 0 only via a *uniform* branch (TDM ignores
//   EXEC, so predication would duplicate the DMA 8x). Next-step X tile is
//   prefetched (global_prefetch_b8). One monotonic global atomic barrier
//   per step (ping-pong buffers make a single barrier sufficient).
// ---------------------------------------------------------------------------

typedef __attribute__((ext_vector_type(16))) __bf16        v16bf;
typedef __attribute__((ext_vector_type(8)))  float         v8f;
typedef __attribute__((ext_vector_type(4)))  unsigned int  v4ui;
typedef __attribute__((ext_vector_type(8)))  int           v8si;
typedef __attribute__((ext_vector_type(4)))  int           v4si;

#define B_    128
#define T_    1024
#define VEC_  512
#define TS_   256
#define U_    512
#define NWG   32          // persistent workgroups
#define NCOL  16          // U_ / NWG columns per WG
#define THREADS 256       // 8 waves

// LDS layout (in ushorts). All bf16 stored as raw ushort.
#define L_WDT 0                       // [NCOL][TS_]  W_d tile, N-major
#define L_WZT (L_WDT + NCOL*TS_)      // [NCOL][TS_]  W_z tile
#define L_UDT (L_WZT + NCOL*TS_)      // [NCOL][U_]   U_d tile
#define L_UZT (L_UDT + NCOL*U_)       // [NCOL][U_]   U_z tile
#define L_SX  (L_UZT + NCOL*U_)       // [B_][TS_]    X_t  (row-major)
#define L_SH  (L_SX  + B_*TS_)        // [B_][U_]     H    (row-major)
#define L_TOT (L_SH  + B_*U_)         // = 122880 ushorts = 245760 bytes

#if defined(__has_builtin)
#if __has_builtin(__builtin_amdgcn_tensor_load_to_lds) && \
    __has_builtin(__builtin_amdgcn_s_wait_tensorcnt)
#define HAVE_TDM 1
#endif
#endif

__device__ __forceinline__ unsigned short f2bf(float x) {
    unsigned int u = __float_as_uint(x);
    u += 0x7fffu + ((u >> 16) & 1u);          // round-to-nearest-even
    return (unsigned short)(u >> 16);
}
__device__ __forceinline__ float softsign(float x) { return x / (1.0f + fabsf(x)); }

// ---------------------------------------------------------------------------
// Phase 1: H0 / pd / pz. 256 blocks x 256 thr; block = (b, u-half).
// ---------------------------------------------------------------------------
__global__ __launch_bounds__(256)
void sghu_init_kernel(const float* __restrict__ Xv, const float* __restrict__ W0,
                      const float* __restrict__ b0, const float* __restrict__ Pd,
                      const float* __restrict__ bd, const float* __restrict__ Pz,
                      const float* __restrict__ bz,
                      float* __restrict__ pd, float* __restrict__ pz,
                      float* __restrict__ H32, unsigned short* __restrict__ Hbf0) {
    __shared__ float sx[VEC_];
    const int b = blockIdx.x >> 1;
    const int u = ((blockIdx.x & 1) << 8) + threadIdx.x;
    sx[threadIdx.x]       = Xv[b * VEC_ + threadIdx.x];
    sx[threadIdx.x + 256] = Xv[b * VEC_ + threadIdx.x + 256];
    __syncthreads();
    float a0 = 0.f, ad = 0.f, az = 0.f;
    #pragma unroll 4
    for (int k = 0; k < VEC_; ++k) {
        const float x = sx[k];
        a0 = fmaf(x, W0[k * U_ + u], a0);
        ad = fmaf(x, Pd[k * U_ + u], ad);
        az = fmaf(x, Pz[k * U_ + u], az);
    }
    const float h0 = softsign(a0 + b0[u]);
    H32[b * U_ + u]  = h0;
    Hbf0[b * U_ + u] = f2bf(h0);
    pd[b * U_ + u] = ad + bd[u];
    pz[b * U_ + u] = az + bz[u];
}

// ---------------------------------------------------------------------------
// Fragment loaders (layouts per CDNA5 ISA §7.12.2, wave32 bf16).
// A (16x32 MxK): lane m=lane&15; a[0..7] = K(kt*32 + hi*8 + 0..7),
//                a[8..15] = K(kt*32 + 16 + hi*8 + 0..7).
// B (32x16 KxN), stored N-major Bt[n][k]: b[0..7] = K(kt*32 + hi*8 + 0..7),
//                b[8..15] = K(kt*32 + 16 + hi*8 + 0..7).
// ---------------------------------------------------------------------------
union Frag { v16bf v; uint4 q[2]; };

__device__ __forceinline__ v16bf ldA(const unsigned short* s, int row, int K,
                                     int kt, int hi) {
    Frag f;
    const unsigned short* p = s + row * K + kt * 32 + hi * 8;
    f.q[0] = *(const uint4*)(p);
    f.q[1] = *(const uint4*)(p + 16);
    return f.v;
}
__device__ __forceinline__ v16bf ldB(const unsigned short* s, int ncol, int K,
                                     int kt, int hi) {
    Frag f;
    const unsigned short* p = s + ncol * K + kt * 32 + hi * 8;
    f.q[0] = *(const uint4*)(p);
    f.q[1] = *(const uint4*)(p + 16);
    return f.v;
}
__device__ __forceinline__ v8f wmma_bf16(v16bf a, v16bf b, v8f c) {
    return __builtin_amdgcn_wmma_f32_16x16x32_bf16(
        false, a, false, b, (short)0, c, false, false);
}

// ---------------------------------------------------------------------------
// Phase 2: persistent recurrence. Grid = NWG, block = 256.
// ---------------------------------------------------------------------------
__global__ __launch_bounds__(THREADS, 1)
void sghu_recur_kernel(const float* __restrict__ Xts,
                       const float* __restrict__ Wd, const float* __restrict__ Ud,
                       const float* __restrict__ Wz, const float* __restrict__ Uz,
                       const float* __restrict__ pd, const float* __restrict__ pz,
                       const float* __restrict__ H32,
                       unsigned short* __restrict__ Hbf,   // 2 * B_*U_ ping-pong
                       unsigned int* __restrict__ barrier_cnt,
                       float* __restrict__ out) {
    extern __shared__ unsigned short smem[];
    unsigned short* sWdT = smem + L_WDT;
    unsigned short* sWzT = smem + L_WZT;
    unsigned short* sUdT = smem + L_UDT;
    unsigned short* sUzT = smem + L_UZT;
    unsigned short* sX   = smem + L_SX;
#if !defined(HAVE_TDM)
    unsigned short* sH   = smem + L_SH;
#else
    const unsigned short* sH = smem + L_SH;   // written only by the TDM
#endif

    const int tid  = threadIdx.x;
    const int lane = tid & 31;
    const int wv   = tid >> 5;      // wave id = M-tile id (8 waves, 128 rows)
    const int hi   = lane >> 4;
    const int lo   = lane & 15;
    const int c0   = blockIdx.x * NCOL;   // column base of this WG

    // Uniform (SGPR) per-wave predicate: TDM must be guarded by a *scalar*
    // branch, not EXEC predication, since tensor ops issue regardless of EXEC.
    const bool wave0 = ((__builtin_amdgcn_readfirstlane(tid) >> 5) == 0);

    // --- preload weight tiles (bf16, N-major) — once for all 1024 steps ---
    for (int idx = tid; idx < NCOL * TS_; idx += THREADS) {
        const int k = idx >> 4, nl = idx & 15;
        sWdT[nl * TS_ + k] = f2bf(Wd[k * U_ + c0 + nl]);
        sWzT[nl * TS_ + k] = f2bf(Wz[k * U_ + c0 + nl]);
    }
    for (int idx = tid; idx < NCOL * U_; idx += THREADS) {
        const int k = idx >> 4, nl = idx & 15;
        sUdT[nl * U_ + k] = f2bf(Ud[k * U_ + c0 + nl]);
        sUzT[nl * U_ + k] = f2bf(Uz[k * U_ + c0 + nl]);
    }

    // --- per-lane persistent state: D-tile layout m = wv*16 + hi*8 + r, n = c0+lo
    v8f h, pdr, pzr;
    #pragma unroll
    for (int r = 0; r < 8; ++r) {
        const int m = wv * 16 + hi * 8 + r;
        h[r]   = H32[m * U_ + c0 + lo];
        pdr[r] = pd [m * U_ + c0 + lo];
        pzr[r] = pz [m * U_ + c0 + lo];
    }
    __syncthreads();

#if defined(HAVE_TDM)
    // Constant part of the Tensor DMA descriptor (CDNA5 ISA):
    // 1-D copy of 16384 x 8-byte elements (128 KB): tile_dim0 = tensor_dim0 =
    // tensor_dim0_stride = 16384, data_size code 3 (8B), no multicast/pad.
    const v8si tdm_g1 = { (int)0x00030000,          // wg_mask=0 | data_size=3
                          (int)(16384u << 16),      // tensor_dim0[15:0]
                          (int)(1u     << 16),      // dim0[31:16]=0 | tensor_dim1=1
                          (int)(16384u << 16),      // dim1[31:16]=0 | tile_dim0
                          0,                        // tile_dim1=0, tile_dim2=0
                          16384, 0, 0 };            // tensor_dim0_stride
    const v4si tdm_z4 = { 0, 0, 0, 0 };
#if __clang_major__ >= 23
    const v8si tdm_z8 = { 0, 0, 0, 0, 0, 0, 0, 0 };
#endif
#endif

    const int n = c0 + lo;
    for (int t = 0; t < T_; ++t) {
        const unsigned short* Hsrc = Hbf + (size_t)(t & 1) * (B_ * U_);
        unsigned short*       Hdst = Hbf + (size_t)((t + 1) & 1) * (B_ * U_);

#if defined(HAVE_TDM)
        // --- wave 0 kicks the TDM: Hbf[t&1] (global) -> sH (LDS), async ---
        if (wave0) {
            const unsigned long long ga = (unsigned long long)(uintptr_t)Hsrc;
            const v4ui g0 = { 1u,                               // count=1, load
                              (unsigned int)(L_SH * 2),         // lds_addr (bytes)
                              (unsigned int)(ga & 0xffffffffu),
                              (unsigned int)((ga >> 32) & 0x01ffffffu)
                                  | (2u << 30) };               // type=2 (image)
#if __clang_major__ >= 23
            __builtin_amdgcn_tensor_load_to_lds(g0, tdm_g1, tdm_z4, tdm_z4,
                                                tdm_z8, 0);
#else
            __builtin_amdgcn_tensor_load_to_lds(g0, tdm_g1, tdm_z4, tdm_z4, 0);
#endif
        }
#endif
        // --- all waves stage X_t (fp32 -> bf16), overlapped with the TDM ---
        for (int idx = tid; idx < (B_ * TS_) / 4; idx += THREADS) {
            const int b = idx >> 6, q = idx & 63;
            const float4 xv = ((const float4*)Xts)[((size_t)b * T_ + t) * 64 + q];
            ushort4 us;
            us.x = f2bf(xv.x); us.y = f2bf(xv.y);
            us.z = f2bf(xv.z); us.w = f2bf(xv.w);
            *(ushort4*)(sX + (size_t)idx * 4) = us;
        }
#if !defined(HAVE_TDM)
        for (int idx = tid; idx < (B_ * U_) / 8; idx += THREADS) {
            ((uint4*)sH)[idx] = ((const uint4*)Hsrc)[idx];
        }
#endif
        // --- prefetch next step's X tile (streams from HBM), 1 per 64B ---
        if (t + 1 < T_) {
            for (int p = tid; p < 2048; p += THREADS) {
                const int b = p >> 4, q = (p & 15) * 4;
                __builtin_prefetch(
                    &((const float4*)Xts)[((size_t)b * T_ + (t + 1)) * 64 + q],
                    0, 0);
            }
        }
#if defined(HAVE_TDM)
        if (wave0) __builtin_amdgcn_s_wait_tensorcnt(0);
#endif
        __syncthreads();   // wave 0 arrives only after the DMA completed

        // --- matmuls: acc = X_t@W + H@U, both gates; A-frags shared; two
        //     independent accumulator chains per gate for load/WMMA overlap ---
        v8f ad0 = {}, ad1 = {}, az0 = {}, az1 = {};
        const int arow = wv * 16 + lo;
        #pragma unroll
        for (int kt = 0; kt < TS_ / 32; kt += 2) {         // 8 K-tiles
            const v16bf a0 = ldA(sX, arow, TS_, kt,     hi);
            const v16bf a1 = ldA(sX, arow, TS_, kt + 1, hi);
            ad0 = wmma_bf16(a0, ldB(sWdT, lo, TS_, kt,     hi), ad0);
            az0 = wmma_bf16(a0, ldB(sWzT, lo, TS_, kt,     hi), az0);
            ad1 = wmma_bf16(a1, ldB(sWdT, lo, TS_, kt + 1, hi), ad1);
            az1 = wmma_bf16(a1, ldB(sWzT, lo, TS_, kt + 1, hi), az1);
        }
        #pragma unroll
        for (int kt = 0; kt < U_ / 32; kt += 2) {          // 16 K-tiles
            const v16bf a0 = ldA(sH, arow, U_, kt,     hi);
            const v16bf a1 = ldA(sH, arow, U_, kt + 1, hi);
            ad0 = wmma_bf16(a0, ldB(sUdT, lo, U_, kt,     hi), ad0);
            az0 = wmma_bf16(a0, ldB(sUzT, lo, U_, kt,     hi), az0);
            ad1 = wmma_bf16(a1, ldB(sUdT, lo, U_, kt + 1, hi), ad1);
            az1 = wmma_bf16(a1, ldB(sUzT, lo, U_, kt + 1, hi), az1);
        }

        // --- gate + state update (fp32 in registers), publish bf16 H ---
        #pragma unroll
        for (int r = 0; r < 8; ++r) {
            const float delta = softsign(pdr[r] + ad0[r] + ad1[r]);
            const float z =
                0.5f + 0.5f * softsign(4.0f * (pzr[r] + az0[r] + az1[r]));
            const float hn = (1.0f - z) * h[r] + z * delta;
            h[r] = hn;
            const int m = wv * 16 + hi * 8 + r;
            Hdst[m * U_ + n] = f2bf(hn);
        }
        if (t == T_ - 1) {
            #pragma unroll
            for (int r = 0; r < 8; ++r) {
                const int m = wv * 16 + hi * 8 + r;
                out[m * U_ + n] = h[r];
            }
        }

        // --- one grid barrier per step (ping-pong makes a single one safe) ---
        __threadfence();
        __syncthreads();
        if (tid == 0) {
            __hip_atomic_fetch_add(barrier_cnt, 1u, __ATOMIC_ACQ_REL,
                                   __HIP_MEMORY_SCOPE_AGENT);
            const unsigned int target = (unsigned int)(t + 1) * NWG;
            while (__hip_atomic_load(barrier_cnt, __ATOMIC_ACQUIRE,
                                     __HIP_MEMORY_SCOPE_AGENT) < target) {
                __builtin_amdgcn_s_sleep(8);
            }
        }
        __syncthreads();
    }
}

// ---------------------------------------------------------------------------
// Launch. Inputs (setup_inputs order):
//  0 X_vecs 1 X_ts 2 W_0 3 b_0 4 P_d 5 W_d 6 U_d 7 b_d 8 P_z 9 W_z 10 U_z 11 b_z
// ---------------------------------------------------------------------------
extern "C" void kernel_launch(void* const* d_in, const int* in_sizes, int n_in,
                              void* d_out, int out_size, void* d_ws, size_t ws_size,
                              hipStream_t stream) {
    const float* Xv  = (const float*)d_in[0];
    const float* Xts = (const float*)d_in[1];
    const float* W0  = (const float*)d_in[2];
    const float* b0  = (const float*)d_in[3];
    const float* Pd  = (const float*)d_in[4];
    const float* Wd  = (const float*)d_in[5];
    const float* Ud  = (const float*)d_in[6];
    const float* bd  = (const float*)d_in[7];
    const float* Pz  = (const float*)d_in[8];
    const float* Wz  = (const float*)d_in[9];
    const float* Uz  = (const float*)d_in[10];
    const float* bz  = (const float*)d_in[11];

    // workspace layout
    unsigned int*   cnt  = (unsigned int*)d_ws;
    float*          pd   = (float*)((char*)d_ws + 256);
    float*          pz   = pd  + B_ * U_;
    float*          H32  = pz  + B_ * U_;
    unsigned short* Hbf  = (unsigned short*)(H32 + B_ * U_);   // 2 buffers

    (void)in_sizes; (void)n_in; (void)out_size; (void)ws_size;

    hipMemsetAsync(d_ws, 0, 256, stream);   // zero barrier counter (capturable)

    sghu_init_kernel<<<B_ * 2, 256, 0, stream>>>(
        Xv, W0, b0, Pd, bd, Pz, bz, pd, pz, H32, Hbf);

    const size_t lds_bytes = (size_t)L_TOT * sizeof(unsigned short); // 245760
    hipFuncSetAttribute((const void*)sghu_recur_kernel,
                        hipFuncAttributeMaxDynamicSharedMemorySize,
                        (int)lds_bytes);  // idempotent host-side attribute
    sghu_recur_kernel<<<NWG, THREADS, lds_bytes, stream>>>(
        Xts, Wd, Ud, Wz, Uz, pd, pz, H32, Hbf, cnt, (float*)d_out);
}
